// EEGDecoder_55405078119339
// MI455X (gfx1250) — matrix operand order repaired and checked by reading, not compiled, not run on hardware
//
#include <hip/hip_runtime.h>

#define T_STEPS 512
#define H1 512
#define H2 256
#define FDIM 64
#define BATCH 256
#define LATENT 256
#define BN_EPS 1e-3f

typedef __attribute__((ext_vector_type(16))) __bf16        v16bf;
typedef __attribute__((ext_vector_type(8)))  float         v8f;
typedef __attribute__((ext_vector_type(4)))  unsigned int  v4u;

// global (address_space(1)) pointer types: keep weight streams as
// global_load_b128 (LOADcnt only) even after laundering.
typedef const __attribute__((address_space(1))) unsigned short* gcus;
typedef const __attribute__((address_space(1))) v4u*            gcv4u;

union FragU { v4u u[2]; v16bf v; };

__device__ static inline unsigned short f2bf(float f) {
  unsigned int u = __float_as_uint(f);
  u += 0x7FFFu + ((u >> 16) & 1u);          // round-to-nearest-even
  return (unsigned short)(u >> 16);
}

__device__ static inline float sigmoidf_fast(float x) {
  return 1.0f / (1.0f + __expf(-x));
}

// Make a pointer opaque to the optimizer (blocks LICM of loads through it
// out of the time loop) while keeping it in SGPRs (saddr addressing).
#define LAUNDER_S(p) asm volatile("" : "+s"(p))

// A-fragment (16x32 bf16, row-major source, row = this lane's M):
// VGPRs 0-3: K = grp*8 + 0..7 ; VGPRs 4-7: K = 16 + grp*8 + 0..7
__device__ static inline v16bf frag_A(const unsigned short* row, int kbase, int grp) {
  FragU f;
  f.u[0] = *(const v4u*)(row + kbase + grp * 8);
  f.u[1] = *(const v4u*)(row + kbase + 16 + grp * 8);
  return f.v;
}

// B-fragment (32x16 bf16) from W^T row-major [N][K] in GLOBAL memory,
// row = this lane's N: VGPRs 0-3: K = grp*8 + 0..7 ; VGPRs 4-7: K = 16 + grp*8 + 0..7
// (two global_load_b128).
__device__ static inline v16bf frag_Bg(gcus row, int kbase, int grp) {
  gcv4u p0 = (gcv4u)(row + kbase + grp * 8);
  gcv4u p1 = (gcv4u)(row + kbase + 16 + grp * 8);
  FragU f;
  f.u[0] = p0[0];
  f.u[1] = p1[0];
  return f.v;
}

__device__ static inline v8f wmma_bf16(v16bf a, v16bf b, v8f c) {
  return __builtin_amdgcn_wmma_f32_16x16x32_bf16(
      /*neg_a=*/false, a, /*neg_b=*/false, b,
      /*c_mod=*/(short)0, c, /*reuse_a=*/false, /*reuse_b=*/false);
}

// ---------------- prep kernels ----------------

// src [K][N] fp32  ->  dst [N][K] bf16
__global__ void transpose_bf16_kernel(unsigned short* __restrict__ dst,
                                      const float* __restrict__ src,
                                      int K, int N) {
  int i = blockIdx.x * blockDim.x + threadIdx.x;
  if (i >= K * N) return;
  int n = i / K;
  int k = i - n * K;
  dst[n * K + k] = f2bf(src[k * N + n]);
}

// bn(x) = x*a + s  with  a = g*rsqrt(v+eps), s = be - m*a
__global__ void bnconst_kernel(float* __restrict__ a, float* __restrict__ s,
                               const float* __restrict__ g, const float* __restrict__ be,
                               const float* __restrict__ m, const float* __restrict__ v,
                               int n) {
  int i = blockIdx.x * blockDim.x + threadIdx.x;
  if (i >= n) return;
  float ai = g[i] * rsqrtf(v[i] + BN_EPS);
  a[i] = ai;
  s[i] = be[i] - m[i] * ai;
}

// s0 = tanh(latent @ W_lat + b_lat);  h0 = s0[:, :H1] (bf16), c0 = s0[:, H1:] (f32)
__global__ void init_state_kernel(const float* __restrict__ latent,
                                  const float* __restrict__ W_lat,
                                  const float* __restrict__ b_lat,
                                  unsigned short* __restrict__ hstate,
                                  float* __restrict__ cstate) {
  int i = blockIdx.x * blockDim.x + threadIdx.x;   // BATCH * 2*H1 threads
  int b = i >> 10;
  int col = i & 1023;
  float acc = b_lat[col];
  const float* lrow = latent + b * LATENT;
  for (int k = 0; k < LATENT; ++k)
    acc += lrow[k] * W_lat[k * (2 * H1) + col];
  float s = tanhf(acc);
  if (col < H1) hstate[b * H1 + col] = f2bf(s);
  else          cstate[b * H1 + (col - H1)] = s;
}

// ---------------- LSTM layer 1 (input is all-zero => xz = b1) ----------------
// grid = BATCH/32 blocks, block = 256 threads (8 waves), M-tile = 32 rows.
// Ping-pong LDS h buffer: one barrier/step, results written immediately.
// Wave w owns hidden tiles {w, w+8, w+16, w+24}.
__global__ __launch_bounds__(256, 2) void lstm1_kernel(
    const unsigned short* __restrict__ Wh1T,   // [4*H1][H1] bf16 (row-major W^T)
    const float* __restrict__ b1,              // [4*H1]
    const unsigned short* __restrict__ hstate, // [BATCH][H1] bf16
    const float* __restrict__ cstate,          // [BATCH][H1] f32
    const float* __restrict__ a1,              // [H1] BN scale
    const float* __restrict__ s1,              // [H1] BN shift
    unsigned short* __restrict__ X1)           // [BATCH][T][H1] bf16 = bn1(h1)
{
  __shared__ __align__(16) unsigned short lds_h[2][32 * H1];   // 2 x 32 KB

  const int tid  = threadIdx.x;
  const int wave = tid >> 5;
  const int lane = tid & 31;
  const int grp  = lane >> 4;
  const int idx  = lane & 15;
  const int b0   = blockIdx.x * 32;

  // stage h0 into read buffer 0
  for (int i = tid; i < 32 * H1; i += 256)
    lds_h[0][i] = hstate[(b0 + (i >> 9)) * H1 + (i & (H1 - 1))];

  // per-(wave,lane) loop invariants
  float c[4][2][8];
  float av[4], sv[4], bias[4][4];
#pragma unroll
  for (int j = 0; j < 4; ++j) {
    const int hb = (wave + 8 * j) * 16;
    av[j] = a1[hb + idx];
    sv[j] = s1[hb + idx];
#pragma unroll
    for (int g = 0; g < 4; ++g) bias[j][g] = b1[g * H1 + hb + idx];
#pragma unroll
    for (int mt = 0; mt < 2; ++mt)
#pragma unroll
      for (int r = 0; r < 8; ++r) {
        int m = mt * 16 + r + grp * 8;
        c[j][mt][r] = cstate[(b0 + m) * H1 + hb + idx];
      }
  }
  __syncthreads();

  gcus whT = (gcus)(unsigned long long)Wh1T;

#pragma unroll 1
  for (int t = 0; t < T_STEPS; ++t) {
    LAUNDER_S(whT);   // opaque each step: weight loads stay IN the loop (L2 stream)

    const unsigned short* hr = lds_h[t & 1];
    unsigned short*       hw = lds_h[(t & 1) ^ 1];

#pragma unroll
    for (int j = 0; j < 4; ++j) {
      const int hb = (wave + 8 * j) * 16;
      v8f z[4][2];
#pragma unroll
      for (int g = 0; g < 4; ++g) {
        const int nb = g * H1 + hb;              // global column of this gate tile
        v8f acc0 = {};
        v8f acc1 = {};
#pragma unroll
        for (int kt = 0; kt < H1 / 32; ++kt) {
          v16bf bf = frag_Bg(whT + (size_t)(nb + idx) * H1, kt * 32, grp);
          acc0 = wmma_bf16(frag_A(hr + idx * H1, kt * 32, grp), bf, acc0);
          acc1 = wmma_bf16(frag_A(hr + (16 + idx) * H1, kt * 32, grp), bf, acc1);
        }
#pragma unroll
        for (int r = 0; r < 8; ++r) { acc0[r] += bias[j][g]; acc1[r] += bias[j][g]; }
        z[g][0] = acc0;
        z[g][1] = acc1;
      }
      // pointwise gates -> write h_t immediately (other buffer, no hazard)
#pragma unroll
      for (int mt = 0; mt < 2; ++mt)
#pragma unroll
        for (int r = 0; r < 8; ++r) {
          float ig = sigmoidf_fast(z[0][mt][r]);
          float fg = sigmoidf_fast(z[1][mt][r]);
          float gg = tanhf(z[2][mt][r]);
          float og = sigmoidf_fast(z[3][mt][r]);
          float cc = fg * c[j][mt][r] + ig * gg;
          c[j][mt][r] = cc;
          float hv = og * tanhf(cc);
          const int m = mt * 16 + r + grp * 8;
          hw[m * H1 + hb + idx] = f2bf(hv);
          X1[((size_t)(b0 + m) * T_STEPS + t) * H1 + hb + idx] =
              f2bf(hv * av[j] + sv[j]);
        }
    }
    __syncthreads();   // h_t complete; next step reads the buffer just written
  }
}

// ---------------- LSTM layer 2 + BN2 + out-projection + tanh ----------------
// z_t = b2 + x_t @ Wx2 (K=512, from X1) + h @ Wh2 (K=256, from LDS)
// M-tile = 32 rows, ping-pong LDS, one barrier/step. Wave w owns hidden tiles
// {w, w+8}. After the barrier, waves 0..3 compute
// out_t = tanh(bn2(h2) @ W_out + b_out), 16 cols x 2 M-tiles each.
__global__ __launch_bounds__(256, 2) void lstm2_kernel(
    const unsigned short* __restrict__ Wx2T,   // [4*H2][H1] bf16
    const unsigned short* __restrict__ Wh2T,   // [4*H2][H2] bf16
    const float* __restrict__ b2,              // [4*H2]
    const float* __restrict__ a2,              // [H2]
    const float* __restrict__ s2,              // [H2]
    const unsigned short* __restrict__ WoutT,  // [FDIM][H2] bf16
    const float* __restrict__ b_out,           // [FDIM]
    const unsigned short* __restrict__ X1,     // [BATCH][T][H1] bf16
    float* __restrict__ out)                   // [BATCH][T][FDIM] f32
{
  __shared__ __align__(16) unsigned short lds_h[2][32 * H2];    // 2 x 16 KB
  __shared__ __align__(16) unsigned short lds_hbn[2][32 * H2];  // 2 x 16 KB

  const int tid  = threadIdx.x;
  const int wave = tid >> 5;
  const int lane = tid & 31;
  const int grp  = lane >> 4;
  const int idx  = lane & 15;
  const int b0   = blockIdx.x * 32;

  for (int i = tid; i < 32 * H2; i += 256) lds_h[0][i] = 0;

  float c[2][2][8] = {};
  float av[2], sv[2], bias[2][4];
#pragma unroll
  for (int j = 0; j < 2; ++j) {
    const int hb = (wave + 8 * j) * 16;
    av[j] = a2[hb + idx];
    sv[j] = s2[hb + idx];
#pragma unroll
    for (int g = 0; g < 4; ++g) bias[j][g] = b2[g * H2 + hb + idx];
  }
  const float bo = (wave < 4) ? b_out[wave * 16 + idx] : 0.0f;
  __syncthreads();

  gcus wxT = (gcus)(unsigned long long)Wx2T;
  gcus whT = (gcus)(unsigned long long)Wh2T;
  gcus woT = (gcus)(unsigned long long)WoutT;

#pragma unroll 1
  for (int t = 0; t < T_STEPS; ++t) {
    LAUNDER_S(wxT);   // keep all weight streams inside the loop
    LAUNDER_S(whT);
    LAUNDER_S(woT);

    const int rd = t & 1;
    const unsigned short* hr  = lds_h[rd];
    unsigned short*       hw  = lds_h[rd ^ 1];
    unsigned short*       hbw = lds_hbn[rd ^ 1];

    // this lane's A rows of x_t for the two M-tiles
    const unsigned short* xrow0 = X1 + ((size_t)(b0 + idx) * T_STEPS + t) * H1;
    const unsigned short* xrow1 = X1 + ((size_t)(b0 + 16 + idx) * T_STEPS + t) * H1;

#pragma unroll
    for (int j = 0; j < 2; ++j) {
      const int hb = (wave + 8 * j) * 16;
      v8f z[4][2];
#pragma unroll
      for (int g = 0; g < 4; ++g) {
        const int nb = g * H2 + hb;
        v8f acc0 = {};
        v8f acc1 = {};
        // x_t @ Wx2  (K = 512)
#pragma unroll
        for (int kt = 0; kt < H1 / 32; ++kt) {
          v16bf bf = frag_Bg(wxT + (size_t)(nb + idx) * H1, kt * 32, grp);
          acc0 = wmma_bf16(frag_A(xrow0, kt * 32, grp), bf, acc0);
          acc1 = wmma_bf16(frag_A(xrow1, kt * 32, grp), bf, acc1);
        }
        // h @ Wh2  (K = 256)
#pragma unroll
        for (int kt = 0; kt < H2 / 32; ++kt) {
          v16bf bf = frag_Bg(whT + (size_t)(nb + idx) * H2, kt * 32, grp);
          acc0 = wmma_bf16(frag_A(hr + idx * H2, kt * 32, grp), bf, acc0);
          acc1 = wmma_bf16(frag_A(hr + (16 + idx) * H2, kt * 32, grp), bf, acc1);
        }
#pragma unroll
        for (int r = 0; r < 8; ++r) { acc0[r] += bias[j][g]; acc1[r] += bias[j][g]; }
        z[g][0] = acc0;
        z[g][1] = acc1;
      }
#pragma unroll
      for (int mt = 0; mt < 2; ++mt)
#pragma unroll
        for (int r = 0; r < 8; ++r) {
          float ig = sigmoidf_fast(z[0][mt][r]);
          float fg = sigmoidf_fast(z[1][mt][r]);
          float gg = tanhf(z[2][mt][r]);
          float og = sigmoidf_fast(z[3][mt][r]);
          float cc = fg * c[j][mt][r] + ig * gg;
          c[j][mt][r] = cc;
          float hv = og * tanhf(cc);
          const int m = mt * 16 + r + grp * 8;
          hw[m * H2 + hb + idx]  = f2bf(hv);
          hbw[m * H2 + hb + idx] = f2bf(hv * av[j] + sv[j]);
        }
    }
    __syncthreads();   // h_t / bn2(h_t) complete in write buffer

    if (wave < 4) {    // out_t = tanh(bn2(h2) @ W_out + b_out); 4 tiles of 16 cols
      const int nb = wave * 16;
      v8f acc0 = {};
      v8f acc1 = {};
#pragma unroll
      for (int kt = 0; kt < H2 / 32; ++kt) {
        v16bf bf = frag_Bg(woT + (size_t)(nb + idx) * H2, kt * 32, grp);
        acc0 = wmma_bf16(frag_A(hbw + idx * H2, kt * 32, grp), bf, acc0);
        acc1 = wmma_bf16(frag_A(hbw + (16 + idx) * H2, kt * 32, grp), bf, acc1);
      }
#pragma unroll
      for (int r = 0; r < 8; ++r) {
        const int m0 = r + grp * 8;
        out[((size_t)(b0 + m0) * T_STEPS + t) * FDIM + nb + idx] = tanhf(acc0[r] + bo);
        out[((size_t)(b0 + 16 + m0) * T_STEPS + t) * FDIM + nb + idx] = tanhf(acc1[r] + bo);
      }
    }
    // safe without extra barrier: step t+1 writes the *other* buffers, and any
    // wave's t+1 writes happen after its own out-proj(t) and after barrier(t).
  }
}

// ---------------- host ----------------

extern "C" void kernel_launch(void* const* d_in, const int* in_sizes, int n_in,
                              void* d_out, int out_size, void* d_ws, size_t ws_size,
                              hipStream_t stream) {
  (void)in_sizes; (void)n_in; (void)out_size;

  const float* latent = (const float*)d_in[0];
  const float* W_lat  = (const float*)d_in[1];
  const float* b_lat  = (const float*)d_in[2];
  /* d_in[3] = Wx1 is dead: LSTM1 input is identically zero */
  const float* Wh1    = (const float*)d_in[4];
  const float* b1     = (const float*)d_in[5];
  const float* g1     = (const float*)d_in[6];
  const float* be1    = (const float*)d_in[7];
  const float* m1     = (const float*)d_in[8];
  const float* v1     = (const float*)d_in[9];
  const float* Wx2    = (const float*)d_in[10];
  const float* Wh2    = (const float*)d_in[11];
  const float* b2     = (const float*)d_in[12];
  const float* g2     = (const float*)d_in[13];
  const float* be2    = (const float*)d_in[14];
  const float* m2     = (const float*)d_in[15];
  const float* v2     = (const float*)d_in[16];
  const float* W_out  = (const float*)d_in[17];
  const float* b_out  = (const float*)d_in[18];

  char* ws = (char*)d_ws;
  size_t oWh1T  = 0;                                         // [2048][512] bf16
  size_t oWh2T  = oWh1T  + (size_t)4 * H1 * H1 * 2;          // [1024][256] bf16
  size_t oWx2T  = oWh2T  + (size_t)4 * H2 * H2 * 2;          // [1024][512] bf16
  size_t oWoutT = oWx2T  + (size_t)4 * H2 * H1 * 2;          // [64][256]   bf16
  size_t oA1    = oWoutT + (size_t)FDIM * H2 * 2;
  size_t oS1    = oA1    + (size_t)H1 * 4;
  size_t oA2    = oS1    + (size_t)H1 * 4;
  size_t oS2    = oA2    + (size_t)H2 * 4;
  size_t oH0    = oS2    + (size_t)H2 * 4;                   // [256][512] bf16
  size_t oC0    = oH0    + (size_t)BATCH * H1 * 2;           // [256][512] f32
  size_t oX1    = oC0    + (size_t)BATCH * H1 * 4;           // [256][512][512] bf16
  size_t total  = oX1    + (size_t)BATCH * T_STEPS * H1 * 2;
  if (ws_size < total) return;

  unsigned short* Wh1T  = (unsigned short*)(ws + oWh1T);
  unsigned short* Wh2T  = (unsigned short*)(ws + oWh2T);
  unsigned short* Wx2T  = (unsigned short*)(ws + oWx2T);
  unsigned short* WoutT = (unsigned short*)(ws + oWoutT);
  float*          A1    = (float*)(ws + oA1);
  float*          S1    = (float*)(ws + oS1);
  float*          A2    = (float*)(ws + oA2);
  float*          S2    = (float*)(ws + oS2);
  unsigned short* H0    = (unsigned short*)(ws + oH0);
  float*          C0    = (float*)(ws + oC0);
  unsigned short* X1    = (unsigned short*)(ws + oX1);

  auto cdiv = [](int a, int b) { return (a + b - 1) / b; };

  // weight transposes to bf16 [N][K]
  transpose_bf16_kernel<<<cdiv(H1 * 4 * H1, 256), 256, 0, stream>>>(Wh1T, Wh1, H1, 4 * H1);
  transpose_bf16_kernel<<<cdiv(H2 * 4 * H2, 256), 256, 0, stream>>>(Wh2T, Wh2, H2, 4 * H2);
  transpose_bf16_kernel<<<cdiv(H1 * 4 * H2, 256), 256, 0, stream>>>(Wx2T, Wx2, H1, 4 * H2);
  transpose_bf16_kernel<<<cdiv(H2 * FDIM, 256), 256, 0, stream>>>(WoutT, W_out, H2, FDIM);

  // folded batch-norm constants
  bnconst_kernel<<<cdiv(H1, 256), 256, 0, stream>>>(A1, S1, g1, be1, m1, v1, H1);
  bnconst_kernel<<<cdiv(H2, 256), 256, 0, stream>>>(A2, S2, g2, be2, m2, v2, H2);

  // h0/c0 = tanh(latent @ W_lat + b_lat)
  init_state_kernel<<<cdiv(BATCH * 2 * H1, 256), 256, 0, stream>>>(latent, W_lat, b_lat, H0, C0);

  // persistent recurrences (M-tile = 32 rows per block)
  lstm1_kernel<<<BATCH / 32, 256, 0, stream>>>(Wh1T, b1, H0, C0, A1, S1, X1);
  lstm2_kernel<<<BATCH / 32, 256, 0, stream>>>(Wx2T, Wh2T, b2, A2, S2, WoutT, b_out,
                                               X1, (float*)d_out);
}
